// BiLSTM_CRF_81561428951665
// MI455X (gfx1250) — matrix operand order repaired and checked by reading, not compiled, not run on hardware
//
#include <hip/hip_runtime.h>
#include <hip/hip_bf16.h>

typedef __attribute__((ext_vector_type(16))) _Float16 v16h;
typedef __attribute__((ext_vector_type(8)))  _Float16 v8h;
typedef __attribute__((ext_vector_type(8)))  float    v8f;

#define NEGV   -10000.0f
#define SLEN   512
#define BATCH  64
#define HID    512
#define GDIM   2048   // 4H
#define EPAD   320    // E=300 padded to mult of 32
#define KCAT   832    // 320 + 512
#define NTAG   12
#define NTPAD  16

// ---------------- prep kernels ----------------

__global__ void prep_embed(const int* __restrict__ tokens,
                           const float* __restrict__ W_emb,
                           _Float16* __restrict__ xh) {
    size_t idx = (size_t)blockIdx.x * 256 + threadIdx.x;
    if (idx >= (size_t)SLEN * BATCH * EPAD) return;
    int k = (int)(idx % EPAD);
    size_t sb = idx / EPAD;
    int b = (int)(sb % BATCH);
    int s = (int)(sb / BATCH);
    float v = 0.f;
    if (k < 300) {
        int tok = tokens[b * SLEN + s];           // tokens is [B,S]
        v = W_emb[(size_t)tok * 300 + k];
    }
    xh[idx] = (_Float16)v;
}

__global__ void prep_wcat(const float* __restrict__ Wih_f, const float* __restrict__ Whh_f,
                          const float* __restrict__ Wih_b, const float* __restrict__ Whh_b,
                          _Float16* __restrict__ wcat) {
    size_t idx = (size_t)blockIdx.x * 256 + threadIdx.x;
    if (idx >= (size_t)2 * GDIM * KCAT) return;
    int k = (int)(idx % KCAT);
    int n = (int)((idx / KCAT) % GDIM);
    int dir = (int)(idx / ((size_t)KCAT * GDIM));
    const float* Wih = dir ? Wih_b : Wih_f;
    const float* Whh = dir ? Whh_b : Whh_f;
    float v = 0.f;
    if (k < 300)        v = Wih[(size_t)n * 300 + k];
    else if (k >= EPAD) v = Whh[(size_t)n * HID + (k - EPAD)];
    wcat[idx] = (_Float16)v;
}

__global__ void prep_wout(const float* __restrict__ W_out, _Float16* __restrict__ woutT) {
    int idx = blockIdx.x * 256 + threadIdx.x;     // 16*1024
    int n = idx / (2 * HID);
    int k = idx % (2 * HID);
    float v = (n < NTAG) ? W_out[(size_t)n * (2 * HID) + k] : 0.f;
    woutT[idx] = (_Float16)v;
}

__global__ void prep_state(const float* __restrict__ h0, const float* __restrict__ c0,
                           _Float16* __restrict__ hstate, float* __restrict__ cstate) {
    int idx = blockIdx.x * 256 + threadIdx.x;     // 2*64*512
    hstate[idx] = (_Float16)h0[idx];
    cstate[idx] = c0[idx];
}

// ---------------- recurrent step: gates = [x_t ; h] @ Wcat^T (WMMA) ----------------
// 1024 waves: dir(2) x tiles(4 x 128) of the 64x2048 gate matrix. K = 832 = 26 chunks of 32.

__global__ void lstm_gemm_step(const _Float16* __restrict__ xh,
                               const _Float16* __restrict__ wcat,
                               const _Float16* __restrict__ hstate,
                               float* __restrict__ gates, int t) {
    int lane = threadIdx.x & 31;
    int gw   = blockIdx.x * 8 + (threadIdx.x >> 5);   // 0..1023
    int dir  = gw >> 9;
    int tile = gw & 511;
    int tm = tile >> 7;                               // 0..3   (batch rows)
    int tn = tile & 127;                              // 0..127 (gate cols)
    int m0 = tm * 16, n0 = tn * 16;
    int half = lane >> 4, lr = lane & 15;
    int tIdx = dir ? (SLEN - 1 - t) : t;

    int brow = m0 + lr;                               // A row (batch index) for this lane
    const _Float16* xrow = xh + ((size_t)tIdx * BATCH + brow) * EPAD;
    const _Float16* hrow = hstate + ((size_t)dir * BATCH + brow) * HID;
    const _Float16* wrow = wcat + ((size_t)dir * GDIM + n0 + lr) * KCAT;

    v8f c = {};
    for (int kb = 0; kb < 26; ++kb) {
        int k0 = kb * 32;
        // chunks 0..9 lie fully in the embedding part (k<320), 10..25 fully in h
        const _Float16* asrc = (kb < 10) ? (xrow + k0) : (hrow + (k0 - EPAD));
        v8h lo = *(const v8h*)(asrc + half * 8);
        v8h hi = *(const v8h*)(asrc + 16 + half * 8);
        v16h a;
#pragma unroll
        for (int j = 0; j < 8; ++j) { a[j] = lo[j]; a[8 + j] = hi[j]; }
        v8h blo = *(const v8h*)(wrow + k0 + half * 8);
        v8h bhi = *(const v8h*)(wrow + k0 + 16 + half * 8);
        v16h bmat;
#pragma unroll
        for (int j = 0; j < 8; ++j) { bmat[j] = blo[j]; bmat[8 + j] = bhi[j]; }
        c = __builtin_amdgcn_wmma_f32_16x16x32_f16(false, a, false, bmat,
                                                   (short)0, c, false, false);
    }
    float* gout = gates + (size_t)dir * BATCH * GDIM;
#pragma unroll
    for (int r = 0; r < 8; ++r) {
        int m = m0 + half * 8 + r;
        gout[(size_t)m * GDIM + n0 + lr] = c[r];
    }
}

// ---------------- LSTM cell elementwise ----------------

__global__ void lstm_elem_step(const float* __restrict__ gates,
                               const float* __restrict__ bias_f,
                               const float* __restrict__ bias_b,
                               float* __restrict__ cstate,
                               _Float16* __restrict__ hstate,
                               _Float16* __restrict__ hout, int t) {
    int idx = blockIdx.x * 256 + threadIdx.x;  // 0..65535 = 2*64*512
    int dir = idx >> 15;
    int rem = idx & 32767;
    int b = rem >> 9;
    int j = rem & 511;
    const float* g    = gates + ((size_t)dir * BATCH + b) * GDIM;
    const float* bias = dir ? bias_b : bias_f;
    float iv = g[j]            + bias[j];
    float fv = g[HID + j]      + bias[HID + j];
    float gv = g[2 * HID + j]  + bias[2 * HID + j];
    float ov = g[3 * HID + j]  + bias[3 * HID + j];
    iv = 1.f / (1.f + __expf(-iv));
    fv = 1.f / (1.f + __expf(-fv));
    ov = 1.f / (1.f + __expf(-ov));
    gv = tanhf(gv);
    size_t sidx = ((size_t)dir * BATCH + b) * HID + j;
    float c = fv * cstate[sidx] + iv * gv;
    cstate[sidx] = c;
    float h = ov * tanhf(c);
    hstate[sidx] = (_Float16)h;
    int s = dir ? (SLEN - 1 - t) : t;          // store at original time position
    hout[(((size_t)dir * SLEN + s) * BATCH + b) * HID + j] = (_Float16)h;
}

// ---------------- output projection: feats = [hf;hb] @ Wout^T (WMMA, N padded to 16) ----

__global__ void proj_wmma(const _Float16* __restrict__ hout,
                          const _Float16* __restrict__ woutT,
                          const float* __restrict__ b_out,
                          float* __restrict__ feats) {
    int lane = threadIdx.x & 31;
    int gw   = blockIdx.x * 8 + (threadIdx.x >> 5);  // 0..2047 (M tiles of 32768)
    int m0 = gw * 16;
    int half = lane >> 4, lr = lane & 15;
    const _Float16* hf = hout;
    const _Float16* hb = hout + (size_t)SLEN * BATCH * HID;
    int m = m0 + lr;                                  // m = s*64 + b
    const _Float16* frow = hf + (size_t)m * HID;
    const _Float16* brow = hb + (size_t)m * HID;
    const _Float16* wrow = woutT + (size_t)lr * (2 * HID);

    v8f c = {};
    for (int kb = 0; kb < 32; ++kb) {
        int k0 = kb * 32;
        const _Float16* asrc = (kb < 16) ? (frow + k0) : (brow + (k0 - HID));
        v8h lo = *(const v8h*)(asrc + half * 8);
        v8h hi = *(const v8h*)(asrc + 16 + half * 8);
        v16h a;
#pragma unroll
        for (int j = 0; j < 8; ++j) { a[j] = lo[j]; a[8 + j] = hi[j]; }
        v8h blo = *(const v8h*)(wrow + k0 + half * 8);
        v8h bhi = *(const v8h*)(wrow + k0 + 16 + half * 8);
        v16h bmat;
#pragma unroll
        for (int j = 0; j < 8; ++j) { bmat[j] = blo[j]; bmat[8 + j] = bhi[j]; }
        c = __builtin_amdgcn_wmma_f32_16x16x32_f16(false, a, false, bmat,
                                                   (short)0, c, false, false);
    }
    float bo = (lr < NTAG) ? b_out[lr] : 0.f;
#pragma unroll
    for (int r = 0; r < 8; ++r) {
        int mm = m0 + half * 8 + r;
        feats[(size_t)mm * NTPAD + lr] = c[r] + bo;
    }
}

// ---------------- CRF forward (persistent single workgroup) ----------------

__global__ void crf_forward(const float* __restrict__ feats,
                            const int* __restrict__ lengths,
                            const float* __restrict__ trans,
                            float* __restrict__ out) {
    __shared__ float alpha[BATCH * NTAG];
    __shared__ float tr[NTAG * NTAG];
    __shared__ float lse[BATCH];
    int tid = threadIdx.x;            // 0..767
    int b  = tid / NTAG;
    int nt = tid % NTAG;
    if (tid < NTAG * NTAG) tr[tid] = trans[tid];
    alpha[tid] = (nt == NTAG - 2) ? 0.f : NEGV;   // START = T-2
    int len = lengths[b];
    __syncthreads();
    for (int s = 0; s < SLEN; ++s) {
        float feat = feats[((size_t)s * BATCH + b) * NTPAD + nt];
        float av[NTAG];
        float mx = -3.4e38f;
#pragma unroll
        for (int p = 0; p < NTAG; ++p) {
            float v = alpha[b * NTAG + p] + tr[nt * NTAG + p];
            av[p] = v;
            mx = fmaxf(mx, v);
        }
        float sum = 0.f;
#pragma unroll
        for (int p = 0; p < NTAG; ++p) sum += __expf(av[p] - mx);
        float val = feat + mx + __logf(sum);
        float old = alpha[tid];
        float nv = (s < len) ? val : old;
        __syncthreads();
        alpha[tid] = nv;
        __syncthreads();
    }
    float term = alpha[tid] + tr[(NTAG - 1) * NTAG + nt];   // STOP = T-1
    __syncthreads();
    alpha[tid] = term;
    __syncthreads();
    if (nt == 0) {
        float mx = -3.4e38f;
        for (int p = 0; p < NTAG; ++p) mx = fmaxf(mx, alpha[b * NTAG + p]);
        float sum = 0.f;
        for (int p = 0; p < NTAG; ++p) sum += __expf(alpha[b * NTAG + p] - mx);
        lse[b] = mx + __logf(sum);
    }
    __syncthreads();
    if (tid == 0) {
        float s = 0.f;
        for (int q = 0; q < BATCH; ++q) s += lse[q];
        out[0] = s / (float)BATCH;
    }
}

// ---------------- launch ----------------

extern "C" void kernel_launch(void* const* d_in, const int* in_sizes, int n_in,
                              void* d_out, int out_size, void* d_ws, size_t ws_size,
                              hipStream_t stream) {
    const int*   tokens = (const int*)d_in[0];
    const int*   lengths= (const int*)d_in[1];
    const float* W_emb  = (const float*)d_in[2];
    const float* W_ih_f = (const float*)d_in[3];
    const float* W_hh_f = (const float*)d_in[4];
    const float* b_f    = (const float*)d_in[5];
    const float* W_ih_b = (const float*)d_in[6];
    const float* W_hh_b = (const float*)d_in[7];
    const float* b_b    = (const float*)d_in[8];
    const float* h0     = (const float*)d_in[9];
    const float* c0     = (const float*)d_in[10];
    const float* W_out  = (const float*)d_in[11];
    const float* b_out  = (const float*)d_in[12];
    const float* trans  = (const float*)d_in[13];

    char* ws = (char*)d_ws;
    _Float16* xh     = (_Float16*)(ws + 0);            // 32768*320 f16 = 20,971,520 B
    _Float16* wcat   = (_Float16*)(ws + 20971520);     // 2*2048*832 f16 = 6,815,744 B
    _Float16* woutT  = (_Float16*)(ws + 27787264);     // 16*1024 f16 = 32,768 B
    _Float16* hstate = (_Float16*)(ws + 27820032);     // 2*64*512 f16 = 131,072 B
    float*    cstate = (float*)   (ws + 27951104);     // 2*64*512 f32 = 262,144 B
    float*    gates  = (float*)   (ws + 28213248);     // 2*64*2048 f32 = 1,048,576 B
    _Float16* hout   = (_Float16*)(ws + 29261824);     // 2*512*64*512 f16 = 67,108,864 B
    float*    feats  = (float*)   (ws + 96370688);     // 32768*16 f32 = 2,097,152 B

    prep_embed<<<40960, 256, 0, stream>>>(tokens, W_emb, xh);
    prep_wcat <<<13312, 256, 0, stream>>>(W_ih_f, W_hh_f, W_ih_b, W_hh_b, wcat);
    prep_wout <<<64,    256, 0, stream>>>(W_out, woutT);
    prep_state<<<256,   256, 0, stream>>>(h0, c0, hstate, cstate);

    for (int t = 0; t < SLEN; ++t) {
        lstm_gemm_step<<<128, 256, 0, stream>>>(xh, wcat, hstate, gates, t);
        lstm_elem_step<<<256, 256, 0, stream>>>(gates, b_f, b_b, cstate, hstate, hout, t);
    }

    proj_wmma  <<<256, 256, 0, stream>>>(hout, woutT, b_out, feats);
    crf_forward<<<1,   768, 0, stream>>>(feats, lengths, trans, (float*)d_out);
}
